// SAGE_1803886264469
// MI455X (gfx1250) — matrix operand order repaired and checked by reading, not compiled, not run on hardware
//
#include <hip/hip_runtime.h>

#define N_NODES 100000
#define N_EDGES 800000
#define D 128
#define LN_EPS 1e-5f

typedef __attribute__((ext_vector_type(16))) _Float16 v16h;
typedef __attribute__((ext_vector_type(8)))  _Float16 v8h;
typedef __attribute__((ext_vector_type(4)))  _Float16 v4h;
typedef __attribute__((ext_vector_type(8)))  float    v8f;

// ---------------- degree ----------------
__global__ __launch_bounds__(256) void deg_kernel(const int* __restrict__ dst,
                                                  float* __restrict__ deg) {
  int e = blockIdx.x * blockDim.x + threadIdx.x;
  if (e < N_EDGES) unsafeAtomicAdd(&deg[dst[e]], 1.0f);
}

__global__ __launch_bounds__(256) void invdeg_kernel(const float* __restrict__ deg,
                                                     float* __restrict__ invd) {
  int i = blockIdx.x * blockDim.x + threadIdx.x;
  if (i < N_NODES) invd[i] = 1.0f / fmaxf(deg[i], 1.0f);
}

// ------------- weight pre-pack: f32 [K=128][N=128] x2 -> f16 B fragments -------------
// Fragment frag = ks*8+nt holds 512 halves: slot = lane*16 + j.
// lane = half*16 + nlo ; element = W[(k0 + (j<8 ? half*8 + j : 16 + half*8 + j - 8))*128 + nt*16 + nlo],
// W = Wl for ks<4 (K 0..127), Wr for ks>=4; k0 = (ks&3)*32.
__global__ __launch_bounds__(256) void pack_w_kernel(const float* __restrict__ Wl,
                                                     const float* __restrict__ Wr,
                                                     _Float16* __restrict__ P) {
  int idx = blockIdx.x * blockDim.x + threadIdx.x;   // 0 .. 64*512-1
  if (idx >= 64 * 512) return;
  int frag = idx >> 9;
  int lane = (idx >> 4) & 31;
  int j    = idx & 15;
  int ks = frag >> 3, nt = frag & 7;
  const float* W = (ks < 4) ? Wl : Wr;
  int k0   = (ks & 3) * 32;
  int half = lane >> 4, nlo = lane & 15;
  int k    = k0 + ((j < 8) ? (half * 8 + j) : (16 + half * 8 + (j - 8)));
  P[idx] = (_Float16)W[(size_t)k * D + nt * 16 + nlo];
}

// ------------- edge scatter: agg[dst] += feat[src] (f32 HW atomics) -------------
__global__ __launch_bounds__(256) void agg_kernel_f32(const float* __restrict__ feat,
                                                      const int* __restrict__ src,
                                                      const int* __restrict__ dst,
                                                      float* __restrict__ agg) {
  int t  = blockIdx.x * blockDim.x + threadIdx.x;
  int e  = t >> 5;
  int f4 = (t & 31) << 2;
  if (e < N_EDGES) {
    int s = src[e], d = dst[e];
    const float4 v = *(const float4*)(feat + (size_t)s * D + f4);
    float* p = agg + (size_t)d * D + f4;
    unsafeAtomicAdd(p + 0, v.x);
    unsafeAtomicAdd(p + 1, v.y);
    unsafeAtomicAdd(p + 2, v.z);
    unsafeAtomicAdd(p + 3, v.w);
  }
}

__global__ __launch_bounds__(256) void agg_kernel_f16(const _Float16* __restrict__ feat,
                                                      const int* __restrict__ src,
                                                      const int* __restrict__ dst,
                                                      float* __restrict__ agg) {
  int t  = blockIdx.x * blockDim.x + threadIdx.x;
  int e  = t >> 5;
  int f4 = (t & 31) << 2;
  if (e < N_EDGES) {
    int s = src[e], d = dst[e];
    const v4h v = *(const v4h*)(feat + (size_t)s * D + f4);
    float* p = agg + (size_t)d * D + f4;
    unsafeAtomicAdd(p + 0, (float)v[0]);
    unsafeAtomicAdd(p + 1, (float)v[1]);
    unsafeAtomicAdd(p + 2, (float)v[2]);
    unsafeAtomicAdd(p + 3, (float)v[3]);
  }
}

// ---- A-fragment loaders (16x32 f16; lane half=0: K{0..7,16..23}, half=1: K{8..15,24..31})
__device__ inline v16h loadA_scaled(const float* abase, float scale) {
  float4 u0 = *(const float4*)(abase + 0);
  float4 u1 = *(const float4*)(abase + 4);
  float4 u2 = *(const float4*)(abase + 16);
  float4 u3 = *(const float4*)(abase + 20);
  v16h a;
  a[0]  = (_Float16)(u0.x * scale); a[1]  = (_Float16)(u0.y * scale);
  a[2]  = (_Float16)(u0.z * scale); a[3]  = (_Float16)(u0.w * scale);
  a[4]  = (_Float16)(u1.x * scale); a[5]  = (_Float16)(u1.y * scale);
  a[6]  = (_Float16)(u1.z * scale); a[7]  = (_Float16)(u1.w * scale);
  a[8]  = (_Float16)(u2.x * scale); a[9]  = (_Float16)(u2.y * scale);
  a[10] = (_Float16)(u2.z * scale); a[11] = (_Float16)(u2.w * scale);
  a[12] = (_Float16)(u3.x * scale); a[13] = (_Float16)(u3.y * scale);
  a[14] = (_Float16)(u3.z * scale); a[15] = (_Float16)(u3.w * scale);
  return a;
}

__device__ inline v16h loadA(const float* abase) {
  float4 u0 = *(const float4*)(abase + 0);
  float4 u1 = *(const float4*)(abase + 4);
  float4 u2 = *(const float4*)(abase + 16);
  float4 u3 = *(const float4*)(abase + 20);
  v16h a;
  a[0]  = (_Float16)u0.x; a[1]  = (_Float16)u0.y; a[2]  = (_Float16)u0.z; a[3]  = (_Float16)u0.w;
  a[4]  = (_Float16)u1.x; a[5]  = (_Float16)u1.y; a[6]  = (_Float16)u1.z; a[7]  = (_Float16)u1.w;
  a[8]  = (_Float16)u2.x; a[9]  = (_Float16)u2.y; a[10] = (_Float16)u2.z; a[11] = (_Float16)u2.w;
  a[12] = (_Float16)u3.x; a[13] = (_Float16)u3.y; a[14] = (_Float16)u3.z; a[15] = (_Float16)u3.w;
  return a;
}

__device__ inline v16h loadA(const _Float16* abase) {
  v8h h0 = *(const v8h*)(abase + 0);    // 16B aligned
  v8h h1 = *(const v8h*)(abase + 16);
  v16h a;
  #pragma unroll
  for (int j = 0; j < 8; j++) { a[j] = h0[j]; a[8 + j] = h1[j]; }
  return a;
}

// ------------- fused SAGE conv: out = (agg*invd)@Wl + feat@Wr + b  [+LN+ReLU] ---
// One wave32 per 16-row tile; 8 C tiles cover all 128 output columns; K=256.
template <typename InT, typename OutT>
__global__ __launch_bounds__(256) void sage_gemm_kernel(
    const InT* feat,                  // [N,128] self features (may alias out)
    const float* __restrict__ agg,    // [N,128] neighbor sums (f32)
    const float* __restrict__ invd,   // [N]
    const _Float16* __restrict__ Wp,  // packed B fragments (64 x 512 halves)
    const float* __restrict__ bias,   // [128]
    const float* __restrict__ gamma,  // [128]
    const float* __restrict__ beta,   // [128]
    OutT* out,                        // [N,128]
    int applyLN)
{
  const int lane = threadIdx.x & 31;
  const int wave = threadIdx.x >> 5;
  const int tile = blockIdx.x * 8 + wave;
  const int m_base = tile * 16;
  if (m_base >= N_NODES) return;        // uniform per wave -> EXEC stays all-1s

  const int half = lane >> 4;
  const int nlo  = lane & 15;
  const int arow = m_base + nlo;        // A-matrix row owned by this lane
  const float idg = invd[arow];

  const v16h* Bp = (const v16h*)Wp;     // 32B per lane-slot

  v8f acc[8];
  #pragma unroll
  for (int t = 0; t < 8; t++)
    #pragma unroll
    for (int r = 0; r < 8; r++) acc[t][r] = 0.0f;

  #pragma unroll
  for (int ks = 0; ks < 8; ks++) {
    v16h a;
    if (ks < 4) {
      const float* abase = agg + (size_t)arow * D + ks * 32 + half * 8;
      a = loadA_scaled(abase, idg);
    } else {
      const InT* abase = feat + (size_t)arow * D + (ks - 4) * 32 + half * 8;
      a = loadA(abase);
    }
    #pragma unroll
    for (int nt = 0; nt < 8; nt++) {
      v16h b = Bp[(ks * 8 + nt) * 32 + lane];   // one 32B contiguous load
      acc[nt] = __builtin_amdgcn_wmma_f32_16x16x32_f16(
          false, a, false, b, (short)0, acc[nt], false, false);
    }
  }

  // ---- epilogue: +bias, optional LayerNorm+ReLU ----
  float bb[8], gg[8], be[8];
  #pragma unroll
  for (int t = 0; t < 8; t++) {
    int col = t * 16 + nlo;
    bb[t] = bias[col];
    gg[t] = 1.0f; be[t] = 0.0f;
    if (applyLN) { gg[t] = gamma[col]; be[t] = beta[col]; }
  }

  #pragma unroll
  for (int r = 0; r < 8; r++) {
    const int row = m_base + r + half * 8;   // C layout: VGPR r -> M = r + 8*half
    float vals[8]; float s = 0.0f, ss = 0.0f;
    #pragma unroll
    for (int t = 0; t < 8; t++) {
      float v = acc[t][r] + bb[t];
      vals[t] = v; s += v; ss += v * v;
    }
    if (applyLN) {
      #pragma unroll
      for (int m = 1; m < 16; m <<= 1) {     // row lives in one 16-lane half
        s  += __shfl_xor(s,  m, 32);
        ss += __shfl_xor(ss, m, 32);
      }
      const float mu   = s * (1.0f / D);
      const float var  = ss * (1.0f / D) - mu * mu;
      const float rstd = rsqrtf(var + LN_EPS);
      #pragma unroll
      for (int t = 0; t < 8; t++) {
        float y = (vals[t] - mu) * rstd * gg[t] + be[t];
        out[(size_t)row * D + t * 16 + nlo] = (OutT)fmaxf(y, 0.0f);
      }
    } else {
      #pragma unroll
      for (int t = 0; t < 8; t++)
        out[(size_t)row * D + t * 16 + nlo] = (OutT)vals[t];
    }
  }
}

extern "C" void kernel_launch(void* const* d_in, const int* in_sizes, int n_in,
                              void* d_out, int out_size, void* d_ws, size_t ws_size,
                              hipStream_t stream) {
  const float* x   = (const float*)d_in[0];
  const int*   src = (const int*)d_in[1];
  const int*   dst = (const int*)d_in[2];
  const float* Wl0 = (const float*)d_in[3];
  const float* Wr0 = (const float*)d_in[4];
  const float* b0  = (const float*)d_in[5];
  const float* Wl1 = (const float*)d_in[6];
  const float* Wr1 = (const float*)d_in[7];
  const float* b1  = (const float*)d_in[8];
  const float* Wl2 = (const float*)d_in[9];
  const float* Wr2 = (const float*)d_in[10];
  const float* b2  = (const float*)d_in[11];
  const float* g0  = (const float*)d_in[12];
  const float* be0 = (const float*)d_in[13];
  const float* g1  = (const float*)d_in[14];
  const float* be1 = (const float*)d_in[15];

  char* w = (char*)d_ws;
  float*     agg  = (float*)w;                         w += (size_t)N_NODES * D * 4;
  _Float16*  hA   = (_Float16*)w;                      w += (size_t)N_NODES * D * 2;
  float*     deg  = (float*)w;                         w += (size_t)N_NODES * 4;
  float*     invd = (float*)w;                         w += (size_t)N_NODES * 4;
  _Float16*  P0   = (_Float16*)w;                      w += 64 * 512 * 2;
  _Float16*  P1   = (_Float16*)w;                      w += 64 * 512 * 2;
  _Float16*  P2   = (_Float16*)w;                      w += 64 * 512 * 2;
  float*     outp = (float*)d_out;

  const size_t aggBytes = (size_t)N_NODES * D * sizeof(float);
  const int degBlocks  = (N_EDGES + 255) / 256;
  const int invBlocks  = (N_NODES + 255) / 256;
  const int edgBlocks  = (N_EDGES * 32 + 255) / 256;   // 32 threads/edge
  const int gemmBlocks = ((N_NODES / 16) + 7) / 8;     // 8 waves/block
  const int packBlocks = (64 * 512 + 255) / 256;

  // one-time: degree + packed f16 weights
  hipMemsetAsync(deg, 0, N_NODES * sizeof(float), stream);
  deg_kernel<<<degBlocks, 256, 0, stream>>>(dst, deg);
  invdeg_kernel<<<invBlocks, 256, 0, stream>>>(deg, invd);
  pack_w_kernel<<<packBlocks, 256, 0, stream>>>(Wl0, Wr0, P0);
  pack_w_kernel<<<packBlocks, 256, 0, stream>>>(Wl1, Wr1, P1);
  pack_w_kernel<<<packBlocks, 256, 0, stream>>>(Wl2, Wr2, P2);

  // layer 0: x(f32) -> hA(f16), LN+ReLU
  hipMemsetAsync(agg, 0, aggBytes, stream);
  agg_kernel_f32<<<edgBlocks, 256, 0, stream>>>(x, src, dst, agg);
  sage_gemm_kernel<float, _Float16><<<gemmBlocks, 256, 0, stream>>>(
      x, agg, invd, P0, b0, g0, be0, hA, 1);

  // layer 1: hA(f16) -> hA(f16) in-place, LN+ReLU (rows only read themselves + agg)
  hipMemsetAsync(agg, 0, aggBytes, stream);
  agg_kernel_f16<<<edgBlocks, 256, 0, stream>>>(hA, src, dst, agg);
  sage_gemm_kernel<_Float16, _Float16><<<gemmBlocks, 256, 0, stream>>>(
      hA, agg, invd, P1, b1, g1, be1, hA, 1);

  // layer 2: hA(f16) -> out(f32), no LN
  hipMemsetAsync(agg, 0, aggBytes, stream);
  agg_kernel_f16<<<edgBlocks, 256, 0, stream>>>(hA, src, dst, agg);
  sage_gemm_kernel<_Float16, float><<<gemmBlocks, 256, 0, stream>>>(
      hA, agg, invd, P2, b2, g0, be0, outp, 0);
}
